// FirstLayer_42210938585767
// MI455X (gfx1250) — matrix operand order repaired and checked by reading, not compiled, not run on hardware
//
#include <hip/hip_runtime.h>
#include <hip/hip_bf16.h>
#include <stdint.h>

typedef __attribute__((ext_vector_type(16))) __bf16 v16bf;
typedef __attribute__((ext_vector_type(8)))  __bf16 v8bf;
typedef __attribute__((ext_vector_type(8)))  float  v8f;

#define ISIZE 1024
#define OSIZE 1024
#define GATES 4096   // 4*OSIZE
#define KTOT  2048   // ISIZE+OSIZE
#define BATCH 32
#define SEQ   256
#define NBLK  64

// Dynamic-LDS layout (offsets relative to dynamic base):
//  [0,128K)    Wh slice: 4 n-tiles x 32 k-chunks x 32 lanes x 32B, exact WMMA
//              B-fragment order -> one contiguous 32B read per lane per WMMA
//  [128K,+65K) h staging: 32 rows x 2080B pitch (padded vs 2048 to avoid
//              16-way LDS bank conflicts on row-parallel fragment reads)
//  [ ... ]     sc: 4096 fp32 gate values for the layernorm/pointwise phase
#define LDS_WH_OFF 0u
#define LDS_WH_SZ  131072u            // 4*32*32*32
#define H_PITCH    2080u
#define LDS_H_OFF  (LDS_WH_OFF + LDS_WH_SZ)
#define LDS_H_SZ   (32u * H_PITCH)    // 66560
#define LDS_SC_OFF (LDS_H_OFF + LDS_H_SZ)
#define LDS_SC_SZ  (4096u * 4u)
#define LDS_TOTAL  (LDS_SC_OFF + LDS_SC_SZ)   // ~214 KB of 320 KB/WGP

// ---------------------------------------------------------------------------
// CDNA5 async global->LDS copy (ASYNCcnt path) + wait
// ---------------------------------------------------------------------------
__device__ __forceinline__ unsigned ldsAddr(const void* p) {
    // low 32 bits of an LDS flat address are the group-segment byte address
    return (unsigned)(uintptr_t)p;
}
__device__ __forceinline__ void async_b128(unsigned lds, const void* g) {
    asm volatile("global_load_async_to_lds_b128 %0, %1, off"
                 :: "v"(lds), "v"(g) : "memory");
}
__device__ __forceinline__ void async_b128_o16(unsigned lds, const void* g) {
    // INST_OFFSET applies to both LDS and global address
    asm volatile("global_load_async_to_lds_b128 %0, %1, off offset:16"
                 :: "v"(lds), "v"(g) : "memory");
}
__device__ __forceinline__ void wait_async0() {
    asm volatile("s_wait_asynccnt 0x0" ::: "memory");
}

// Monotonic grid-wide barrier (64 co-resident blocks; counter zeroed per call)
__device__ __forceinline__ void grid_sync(unsigned* bar, unsigned target) {
    __threadfence();
    __syncthreads();
    if (threadIdx.x == 0) {
        __hip_atomic_fetch_add(bar, 1u, __ATOMIC_RELEASE, __HIP_MEMORY_SCOPE_AGENT);
        while (__hip_atomic_load(bar, __ATOMIC_ACQUIRE, __HIP_MEMORY_SCOPE_AGENT) < target) {
            __builtin_amdgcn_s_sleep(8);
        }
    }
    __syncthreads();
}

// ---------------------------------------------------------------------------
// W (2048x4096 fp32 row-major) -> WT (4096x2048 bf16, column-major of W)
// ---------------------------------------------------------------------------
__global__ __launch_bounds__(256) void transpose_w(const float* __restrict__ W,
                                                   __bf16* __restrict__ WT) {
    __shared__ float tile[32][33];
    int nBase = blockIdx.x * 32;
    int kBase = blockIdx.y * 32;
    for (int r = threadIdx.y; r < 32; r += 8)
        tile[r][threadIdx.x] = W[(size_t)(kBase + r) * GATES + nBase + threadIdx.x];
    __syncthreads();
    for (int r = threadIdx.y; r < 32; r += 8)
        WT[(size_t)(nBase + r) * KTOT + kBase + threadIdx.x] =
            (__bf16)tile[threadIdx.x][r];
}

// ---------------------------------------------------------------------------
// h init (broadcast init_hx) + zero the grid-barrier counter
// ---------------------------------------------------------------------------
__global__ __launch_bounds__(256) void init_state(const float* __restrict__ hx,
                                                  __bf16* __restrict__ hbf,
                                                  unsigned* __restrict__ bar) {
    int i = blockIdx.x * 256 + threadIdx.x;
    if (i == 0) *bar = 0u;
    if (i < BATCH * OSIZE) hbf[i] = (__bf16)hx[i & (OSIZE - 1)];
}

// ---------------------------------------------------------------------------
// Input projection: XW[row,j] = x[row,:] @ W[:1024,j] + bias[j], row=b*256+s.
// One wave = 16x64 tile, A reused across 4 WMMAs, software-pipelined so the
// next k-step's loads are in flight while the current WMMAs execute.
// ---------------------------------------------------------------------------
__global__ __launch_bounds__(256) void input_proj(const float* __restrict__ x,
                                                  const __bf16* __restrict__ WT,
                                                  const float* __restrict__ bias,
                                                  float* __restrict__ XW) {
    int lane  = threadIdx.x & 31;
    int w     = blockIdx.x * 8 + (threadIdx.x >> 5);
    int tileM = w % 512;
    int group = w / 512;
    int half  = lane >> 4;
    int mrow  = lane & 15;
    int ncol  = lane & 15;
    int nBase = group * 64;

    const float* xrow = x + (size_t)(tileM * 16 + mrow) * ISIZE;
    const __bf16* wt0 = WT + (size_t)(nBase + ncol) * KTOT;

    v8f acc0 = {0,0,0,0,0,0,0,0};
    v8f acc1 = {0,0,0,0,0,0,0,0};
    v8f acc2 = {0,0,0,0,0,0,0,0};
    v8f acc3 = {0,0,0,0,0,0,0,0};

    float4 f0, f1, f2, f3;
    v16bf nb0, nb1, nb2, nb3;
    auto loadA = [&](int kb) {
        int k0 = kb + half * 16;
        f0 = *(const float4*)(xrow + k0);
        f1 = *(const float4*)(xrow + k0 + 4);
        f2 = *(const float4*)(xrow + k0 + 8);
        f3 = *(const float4*)(xrow + k0 + 12);
    };
    auto loadB = [&](int kb) {
        int kB = kb + half * 16;
        nb0 = *(const v16bf*)(wt0 + (size_t)0  * KTOT + kB);
        nb1 = *(const v16bf*)(wt0 + (size_t)16 * KTOT + kB);
        nb2 = *(const v16bf*)(wt0 + (size_t)32 * KTOT + kB);
        nb3 = *(const v16bf*)(wt0 + (size_t)48 * KTOT + kB);
    };
    loadA(0);
    loadB(0);

    for (int kb = 0; kb < ISIZE; kb += 32) {
        // consume current raw data
        v16bf a;
        a[0]=(__bf16)f0.x;  a[1]=(__bf16)f0.y;  a[2]=(__bf16)f0.z;  a[3]=(__bf16)f0.w;
        a[4]=(__bf16)f1.x;  a[5]=(__bf16)f1.y;  a[6]=(__bf16)f1.z;  a[7]=(__bf16)f1.w;
        a[8]=(__bf16)f2.x;  a[9]=(__bf16)f2.y;  a[10]=(__bf16)f2.z; a[11]=(__bf16)f2.w;
        a[12]=(__bf16)f3.x; a[13]=(__bf16)f3.y; a[14]=(__bf16)f3.z; a[15]=(__bf16)f3.w;
        v16bf b0 = nb0, b1 = nb1, b2 = nb2, b3 = nb3;

        // prefetch next iteration while WMMAs run
        if (kb + 32 < ISIZE) { loadA(kb + 32); loadB(kb + 32); }

        acc0 = __builtin_amdgcn_wmma_f32_16x16x32_bf16(false, a, false, b0, (short)0, acc0, false, false);
        acc1 = __builtin_amdgcn_wmma_f32_16x16x32_bf16(false, a, false, b1, (short)0, acc1, false, false);
        acc2 = __builtin_amdgcn_wmma_f32_16x16x32_bf16(false, a, false, b2, (short)0, acc2, false, false);
        acc3 = __builtin_amdgcn_wmma_f32_16x16x32_bf16(false, a, false, b3, (short)0, acc3, false, false);
    }

    int colBase = nBase + ncol;
#pragma unroll
    for (int r = 0; r < 8; r++) {
        int row = tileM * 16 + r + half * 8;
        float* orow = XW + (size_t)row * GATES;
        orow[colBase]      = acc0[r] + bias[colBase];
        orow[colBase + 16] = acc1[r] + bias[colBase + 16];
        orow[colBase + 32] = acc2[r] + bias[colBase + 32];
        orow[colBase + 48] = acc3[r] + bias[colBase + 48];
    }
}

// ---------------------------------------------------------------------------
// Persistent fused recurrence: 64 blocks x 256 threads (512 tile-waves).
//  prologue: async-load block's 128KB Wh slice into LDS (stays all 256 steps)
//  per step: async-copy h into LDS -> WMMA GEMM (operands from LDS, register
//            double-buffered) -> grid sync -> blocks 0..31 do layernorm+gates
//            (c in registers) -> grid sync
// ---------------------------------------------------------------------------
__global__ __launch_bounds__(256) void lstm_persistent(
        const __bf16* __restrict__ WT, const float* __restrict__ XW,
        const float* __restrict__ gamma, const float* __restrict__ beta,
        const float* __restrict__ cx,
        float* __restrict__ comb, __bf16* __restrict__ hbf,
        float* __restrict__ out, unsigned* __restrict__ bar) {
    extern __shared__ char smem[];
    __shared__ float r1[256];
    __shared__ float r2[256];

    const int tid   = threadIdx.x;
    const int wv    = tid >> 5;
    const int lane  = tid & 31;
    const int n4    = wv >> 1;          // 0..3: which of the block's 4 N tiles
    const int tileM = wv & 1;
    const int tileN = blockIdx.x * 4 + n4;
    const int half  = lane >> 4;
    const int mrow  = lane & 15;
    const int ncol  = lane & 15;

    // ---- prologue: stage recurrent weights (W rows 1024..2047) in fragment order
    {
        const __bf16* gBase = WT + (size_t)(tileN * 16 + ncol) * KTOT + ISIZE + half * 16;
        unsigned lBase = ldsAddr(smem) + LDS_WH_OFF + (unsigned)n4 * 32768u + (unsigned)lane * 32u;
        int kbi0 = (wv & 1) ? 16 : 0;
        for (int kbi = kbi0; kbi < kbi0 + 16; kbi++) {
            async_b128(lBase + (unsigned)kbi * 1024u, gBase + kbi * 32);
            async_b128_o16(lBase + (unsigned)kbi * 1024u, gBase + kbi * 32);
        }
    }
    wait_async0();
    __syncthreads();

    // cell state lives in registers of the 32 pointwise blocks for all 256 steps
    float cReg[4];
    if (blockIdx.x < BATCH) {
#pragma unroll
        for (int i = 0; i < 4; i++) cReg[i] = cx[tid + i * 256];
    }

    const char* hRow = smem + LDS_H_OFF + (unsigned)(tileM * 16 + mrow) * H_PITCH;
    const char* wB   = smem + LDS_WH_OFF + (unsigned)n4 * 32768u + (unsigned)lane * 32u;
    unsigned hLdsBase = ldsAddr(smem) + LDS_H_OFF;

    unsigned phase = 0;
    for (int t = 0; t < SEQ; t++) {
        // ---- async-copy h (32x1024 bf16, 64KB) into LDS with padded pitch
#pragma unroll
        for (int i = 0; i < 16; i++) {
            int ch  = tid + i * 256;         // 0..4095 16B chunks
            int row = ch >> 7, c16 = ch & 127;
            async_b128(hLdsBase + (unsigned)row * H_PITCH + (unsigned)c16 * 16u,
                       hbf + row * OSIZE + c16 * 8);
        }
        wait_async0();
        __syncthreads();

        // ---- 16x16 tile GEMM from LDS, register double-buffered so ds_load
        //      latency overlaps the dependent WMMA accumulate chain
        v8f acc = {0,0,0,0,0,0,0,0};
        v8bf alo = *(const v8bf*)(hRow + (half * 16) * 2);
        v8bf ahi = *(const v8bf*)(hRow + (half * 16) * 2 + 16);
        v16bf bfr = *(const v16bf*)(wB);
#pragma unroll 4
        for (int kbi = 0; kbi < 32; kbi++) {
            v16bf a;
#pragma unroll
            for (int i = 0; i < 8; i++) { a[i] = alo[i]; a[8 + i] = ahi[i]; }
            v16bf bcur = bfr;
            if (kbi < 31) {
                int k1 = (kbi + 1) * 32 + half * 16;
                alo = *(const v8bf*)(hRow + k1 * 2);
                ahi = *(const v8bf*)(hRow + k1 * 2 + 16);
                bfr = *(const v16bf*)(wB + (kbi + 1) * 1024);
            }
            acc = __builtin_amdgcn_wmma_f32_16x16x32_bf16(false, a, false, bcur, (short)0, acc, false, false);
        }
        int col = tileN * 16 + ncol;
#pragma unroll
        for (int r = 0; r < 8; r++) {
            int bidx = tileM * 16 + r + half * 8;   // batch index
            comb[(size_t)bidx * GATES + col] =
                acc[r] + XW[((size_t)bidx * SEQ + t) * GATES + col];
        }

        phase++; grid_sync(bar, phase * NBLK);

        // ---- pointwise: layernorm over 4096 gates + LSTM gate math
        if (blockIdx.x < BATCH) {
            float* scf = (float*)(smem + LDS_SC_OFF);
            const float* cb = comb + (size_t)blockIdx.x * GATES;
            float s = 0.f, s2 = 0.f;
#pragma unroll
            for (int i = 0; i < 16; i++) {
                float v = cb[tid + i * 256];
                scf[tid + i * 256] = v;
                s += v; s2 += v * v;
            }
            r1[tid] = s; r2[tid] = s2;
            __syncthreads();
            for (int off = 128; off > 0; off >>= 1) {
                if (tid < off) { r1[tid] += r1[tid + off]; r2[tid] += r2[tid + off]; }
                __syncthreads();
            }
            float mean = r1[0] * (1.f / GATES);
            float var  = r2[0] * (1.f / GATES) - mean * mean;
            float rstd = rsqrtf(var + 1e-5f);
#pragma unroll
            for (int i = 0; i < 4; i++) {
                int n = tid + i * 256;
                float ig = (scf[n]        - mean) * rstd * gamma[n]        + beta[n];
                float fg = (scf[1024 + n] - mean) * rstd * gamma[1024 + n] + beta[1024 + n];
                float gg = (scf[2048 + n] - mean) * rstd * gamma[2048 + n] + beta[2048 + n];
                float og = (scf[3072 + n] - mean) * rstd * gamma[3072 + n] + beta[3072 + n];
                float si = 1.f / (1.f + expf(-ig));
                float sf = 1.f / (1.f + expf(-fg));
                float so = 1.f / (1.f + expf(-og));
                float cn = sf * cReg[i] + si * tanhf(gg);
                float hn = so * cn;                 // reference: h = sigmoid(og)*c
                cReg[i] = cn;
                hbf[blockIdx.x * OSIZE + n] = (__bf16)hn;
                out[((size_t)blockIdx.x * SEQ + t) * OSIZE + n] = hn;
            }
        }

        phase++; grid_sync(bar, phase * NBLK);
    }
}

extern "C" void kernel_launch(void* const* d_in, const int* in_sizes, int n_in,
                              void* d_out, int out_size, void* d_ws, size_t ws_size,
                              hipStream_t stream) {
    (void)in_sizes; (void)n_in; (void)out_size; (void)ws_size;
    const float* x     = (const float*)d_in[0];
    const float* W     = (const float*)d_in[1];
    const float* bias  = (const float*)d_in[2];
    const float* gamma = (const float*)d_in[3];
    const float* beta  = (const float*)d_in[4];
    const float* hx    = (const float*)d_in[5];
    const float* cx    = (const float*)d_in[6];
    float* out = (float*)d_out;

    char* ws = (char*)d_ws;
    size_t off = 0;
    float* XW = (float*)(ws + off);   off += (size_t)BATCH * SEQ * GATES * sizeof(float);
    __bf16* WT = (__bf16*)(ws + off); off += (size_t)GATES * KTOT * sizeof(__bf16);
    float* comb = (float*)(ws + off); off += (size_t)BATCH * GATES * sizeof(float);
    __bf16* hbf = (__bf16*)(ws + off); off += (size_t)BATCH * OSIZE * sizeof(__bf16);
    off = (off + 255) & ~(size_t)255;
    unsigned* bar = (unsigned*)(ws + off);

    (void)hipFuncSetAttribute((const void*)lstm_persistent,
                              hipFuncAttributeMaxDynamicSharedMemorySize,
                              (int)LDS_TOTAL);

    transpose_w<<<dim3(GATES / 32, KTOT / 32), dim3(32, 8), 0, stream>>>(W, WT);
    init_state<<<(BATCH * OSIZE) / 256, 256, 0, stream>>>(hx, hbf, bar);
    input_proj<<<4096, 256, 0, stream>>>(x, WT, bias, XW);
    lstm_persistent<<<NBLK, 256, LDS_TOTAL, stream>>>(WT, XW, gamma, beta, cx,
                                                      comb, hbf, out, bar);
}
